// IncidenceConv_6227702579796
// MI455X (gfx1250) — compile-verified, matrix operand order checked by reading, not run on hardware
//
#include <hip/hip_runtime.h>
#include <hip/hip_bf16.h>

#define N_S   100000
#define N_T   50000
#define NEDGE 600000
#define DIM   128
#define BN_EPS 1e-5f

typedef __attribute__((ext_vector_type(2)))  float   v2f;
typedef __attribute__((ext_vector_type(8)))  float   v8f;
typedef __attribute__((ext_vector_type(16))) __bf16  v16bf;

// ---------------------------------------------------------------------------
// Kernel 1: H = (1 + eps) * x_t   (vectorized float4, fills the agg buffer)
// ---------------------------------------------------------------------------
__global__ __launch_bounds__(256) void gin_init_h(const float* __restrict__ x_t,
                                                  const float* __restrict__ eps,
                                                  float* __restrict__ H) {
    int i = blockIdx.x * blockDim.x + threadIdx.x;
    const int n4 = N_T * DIM / 4;
    if (i >= n4) return;
    float s = 1.0f + eps[0];
    float4 v = ((const float4*)x_t)[i];
    v.x *= s; v.y *= s; v.z *= s; v.w *= s;
    ((float4*)H)[i] = v;
}

// ---------------------------------------------------------------------------
// Kernel 2: edge scatter-add. One wave per edge; 32 lanes x float4 = 128 cols.
// x_s (51MB) and H (25.6MB) both fit in the 192MB L2 -> atomics resolve in L2.
// ---------------------------------------------------------------------------
__global__ __launch_bounds__(256) void gin_scatter(const float* __restrict__ x_s,
                                                   const int* __restrict__ src,
                                                   const int* __restrict__ dst,
                                                   float* __restrict__ H) {
    int wave = (blockIdx.x * blockDim.x + threadIdx.x) >> 5;
    int lane = threadIdx.x & 31;
    if (wave >= NEDGE) return;
    int s = src[wave];
    int d = dst[wave];
    float4 v = ((const float4*)(x_s + (size_t)s * DIM))[lane];
    float* out = H + (size_t)d * DIM + lane * 4;
    atomicAdd(out + 0, v.x);
    atomicAdd(out + 1, v.y);
    atomicAdd(out + 2, v.z);
    atomicAdd(out + 3, v.w);
}

// ---------------------------------------------------------------------------
// Kernel 3: Z = H @ W^T + bias via WMMA. One wave = one 16x16 output tile.
// Block = 256 threads = 8 waves = all 8 N-tiles of one M-tile (grid = 3125).
// fp32 path: V_WMMA_F32_16X16X4_F32, K-loop of 32 steps.
//   A (16x4 f32): lanes 0-15 -> M=lane, K={kt,kt+1}; lanes 16-31 -> K={kt+2,kt+3}
//   B (4x16 f32): lanes 0-15 -> N=lane, K={kt,kt+1}; lanes 16-31 -> K={kt+2,kt+3}
//   C/D (16x16 f32): VGPR r -> M=r (lanes 0-15) / M=r+8 (lanes 16-31), N=lane&15
// ---------------------------------------------------------------------------
__global__ __launch_bounds__(256) void gin_gemm_bias(const float* __restrict__ H,
                                                     const float* __restrict__ W,
                                                     const float* __restrict__ bias,
                                                     float* __restrict__ Z) {
    const int lane  = threadIdx.x & 31;
    const int ntile = threadIdx.x >> 5;     // 0..7
    const int mtile = blockIdx.x;           // 0..3124

    const int mn  = lane & 15;
    const int hi  = lane >> 4;              // 0 or 1 (K-half)
    const float* hrow = H + (size_t)(mtile * 16 + mn) * DIM;   // A row (M = mn)
    const float* wrow = W + (size_t)(ntile * 16 + mn) * DIM;   // B col: W[n][k] = (W^T)[k][n]

    v8f acc = {};
#if __has_builtin(__builtin_amdgcn_wmma_f32_16x16x4_f32)
    for (int kt = 0; kt < DIM; kt += 4) {
        v2f a, b;
        a.x = hrow[kt + hi * 2 + 0];
        a.y = hrow[kt + hi * 2 + 1];
        b.x = wrow[kt + hi * 2 + 0];
        b.y = wrow[kt + hi * 2 + 1];
        acc = __builtin_amdgcn_wmma_f32_16x16x4_f32(false, a, false, b,
                                                    (short)0, acc, false, false);
    }
#else
    // Fallback: codegen-confirmed bf16 WMMA (GEMM is bandwidth-bound anyway).
    for (int kt = 0; kt < DIM; kt += 32) {
        v16bf a, b;
        #pragma unroll
        for (int j = 0; j < 16; ++j) {
            a[j] = (__bf16)hrow[kt + hi * 16 + j];
            b[j] = (__bf16)wrow[kt + hi * 16 + j];
        }
        acc = __builtin_amdgcn_wmma_f32_16x16x32_bf16(false, a, false, b,
                                                      (short)0, acc, false, false);
    }
#endif
    const int n  = ntile * 16 + (lane & 15);
    const float bv = bias[n];
    const int mbase = mtile * 16 + hi * 8;
    #pragma unroll
    for (int r = 0; r < 8; ++r)
        Z[(size_t)(mbase + r) * DIM + n] = acc[r] + bv;
}

// ---------------------------------------------------------------------------
// Kernel 4: zero the stats accumulators (sum[128] | sumsq[128])
// ---------------------------------------------------------------------------
__global__ void gin_zero_stats(float* __restrict__ sums) {
    if (threadIdx.x < 256) sums[threadIdx.x] = 0.0f;
}

// ---------------------------------------------------------------------------
// Kernel 5: per-column partial sum / sum-of-squares over a row chunk,
// LDS-reduced, one atomicAdd per column per block. Z is L2-resident.
// ---------------------------------------------------------------------------
#define ROWS_PER_BLK 512
__global__ __launch_bounds__(256) void gin_col_stats(const float* __restrict__ Z,
                                                     float* __restrict__ sums) {
    __shared__ float ls[256];
    __shared__ float lq[256];
    const int c    = threadIdx.x & 127;
    const int rofs = threadIdx.x >> 7;      // 0 or 1
    int rbeg = blockIdx.x * ROWS_PER_BLK;
    int rend = rbeg + ROWS_PER_BLK;
    if (rend > N_T) rend = N_T;
    float s = 0.0f, q = 0.0f;
    for (int r = rbeg + rofs; r < rend; r += 2) {
        float v = Z[(size_t)r * DIM + c];
        s += v;
        q += v * v;
    }
    ls[threadIdx.x] = s;
    lq[threadIdx.x] = q;
    __syncthreads();
    if (threadIdx.x < 128) {
        s = ls[threadIdx.x] + ls[threadIdx.x + 128];
        q = lq[threadIdx.x] + lq[threadIdx.x + 128];
        atomicAdd(&sums[c], s);
        atomicAdd(&sums[128 + c], q);
    }
}

// ---------------------------------------------------------------------------
// Kernel 6: finalize BN params: scale = gamma*rsqrt(var+eps), shift = beta-mu*scale
// (biased variance E[z^2]-mu^2, matching jnp.mean of squared deviation)
// ---------------------------------------------------------------------------
__global__ void gin_bn_finalize(const float* __restrict__ sums,
                                const float* __restrict__ gamma,
                                const float* __restrict__ beta,
                                float* __restrict__ ss) {
    int c = threadIdx.x;
    if (c >= 128) return;
    const float inv_n = 1.0f / (float)N_T;
    float mu  = sums[c] * inv_n;
    float var = sums[128 + c] * inv_n - mu * mu;
    float sc  = gamma[c] * rsqrtf(var + BN_EPS);
    ss[c]       = sc;
    ss[128 + c] = beta[c] - mu * sc;
}

// ---------------------------------------------------------------------------
// Kernel 7: O = relu(Z * scale + shift), float4 vectorized
// ---------------------------------------------------------------------------
__global__ __launch_bounds__(256) void gin_bn_relu(const float* __restrict__ Z,
                                                   const float* __restrict__ ss,
                                                   float* __restrict__ O) {
    int i = blockIdx.x * blockDim.x + threadIdx.x;
    const int n4 = N_T * DIM / 4;
    if (i >= n4) return;
    int cb = (i & 31) * 4;                  // column base (0..124)
    float4 z  = ((const float4*)Z)[i];
    float4 sc = *(const float4*)(ss + cb);
    float4 sh = *(const float4*)(ss + 128 + cb);
    float4 o;
    o.x = fmaxf(z.x * sc.x + sh.x, 0.0f);
    o.y = fmaxf(z.y * sc.y + sh.y, 0.0f);
    o.z = fmaxf(z.z * sc.z + sh.z, 0.0f);
    o.w = fmaxf(z.w * sc.w + sh.w, 0.0f);
    ((float4*)O)[i] = o;
}

// ---------------------------------------------------------------------------
extern "C" void kernel_launch(void* const* d_in, const int* in_sizes, int n_in,
                              void* d_out, int out_size, void* d_ws, size_t ws_size,
                              hipStream_t stream) {
    const float* x_s   = (const float*)d_in[0];
    const float* x_t   = (const float*)d_in[1];
    const int*   src   = (const int*)d_in[2];
    const int*   dst   = (const int*)d_in[3];
    const float* eps   = (const float*)d_in[4];
    const float* W1    = (const float*)d_in[5];
    const float* b1    = (const float*)d_in[6];
    const float* g1    = (const float*)d_in[7];
    const float* be1   = (const float*)d_in[8];
    const float* W2    = (const float*)d_in[9];
    const float* b2    = (const float*)d_in[10];
    const float* g2    = (const float*)d_in[11];
    const float* be2   = (const float*)d_in[12];

    float* H    = (float*)d_ws;                       // [N_T * DIM]
    float* Z    = H + (size_t)N_T * DIM;              // [N_T * DIM]
    float* sums = Z + (size_t)N_T * DIM;              // [256]
    float* ss   = sums + 256;                         // [256]

    const int n4       = N_T * DIM / 4;
    const int ew_blks  = (NEDGE * 32 + 255) / 256;    // one wave per edge
    const int gemm_blk = N_T / 16;                    // 3125
    const int st_blks  = (N_T + ROWS_PER_BLK - 1) / ROWS_PER_BLK;

    // Aggregate: H = (1+eps)*x_t + segment_sum(x_s[src], dst)
    gin_init_h <<<(n4 + 255) / 256, 256, 0, stream>>>(x_t, eps, H);
    gin_scatter<<<ew_blks, 256, 0, stream>>>(x_s, src, dst, H);

    // Layer 1: Z = H@W1^T + b1 ; H = relu(bn(Z))
    gin_gemm_bias  <<<gemm_blk, 256, 0, stream>>>(H, W1, b1, Z);
    gin_zero_stats <<<1, 256, 0, stream>>>(sums);
    gin_col_stats  <<<st_blks, 256, 0, stream>>>(Z, sums);
    gin_bn_finalize<<<1, 128, 0, stream>>>(sums, g1, be1, ss);
    gin_bn_relu    <<<(n4 + 255) / 256, 256, 0, stream>>>(Z, ss, H);

    // Layer 2: Z = H@W2^T + b2 ; out = relu(bn(Z))
    gin_gemm_bias  <<<gemm_blk, 256, 0, stream>>>(H, W2, b2, Z);
    gin_zero_stats <<<1, 256, 0, stream>>>(sums);
    gin_col_stats  <<<st_blks, 256, 0, stream>>>(Z, sums);
    gin_bn_finalize<<<1, 128, 0, stream>>>(sums, g2, be2, ss);
    gin_bn_relu    <<<(n4 + 255) / 256, 256, 0, stream>>>(Z, ss, (float*)d_out);
}
